// NodeGraphNet_89060441850434
// MI455X (gfx1250) — compile-verified, matrix operand order checked by reading, not run on hardware
//
#include <hip/hip_runtime.h>
#include <hip/hip_bf16.h>

// ---------------- static problem sizes ----------------
#define NN   100096
#define EE   1601536
#define BB   128
#define NPG  782
#define HH   128
#define CC   10
#define LL   3
#define EPSI 1e-5f

typedef __attribute__((ext_vector_type(16))) __bf16     v16bf;
typedef __attribute__((ext_vector_type(8)))  float      v8f;
typedef __attribute__((ext_vector_type(4)))  unsigned   v4u;
typedef __attribute__((ext_vector_type(4)))  float      v4f;

union ABTile { v4u q[2]; v16bf v; };

__device__ __forceinline__ unsigned short f2bf(float f) {
    unsigned u = __float_as_uint(f);
    unsigned r = (u + 0x7FFFu + ((u >> 16) & 1u)) >> 16;   // RNE
    return (unsigned short)r;
}

// ---------------- utility kernels ----------------
__global__ void k_fill(float* p, size_t n, float v) {
    size_t i = (size_t)blockIdx.x * blockDim.x + threadIdx.x;
    if (i < n) p[i] = v;
}

__global__ void k_pad16(const float* src, float* dst, int nvalid) {
    int i = threadIdx.x;
    if (i < 16) dst[i] = (i < nvalid) ? src[i] : 0.0f;
}

__global__ void k_degrees(const int* src, const int* dst, float* degO, float* degI, int e) {
    int i = blockIdx.x * blockDim.x + threadIdx.x;
    if (i < e) {
        unsafeAtomicAdd(&degO[src[i]], 1.0f);
        unsafeAtomicAdd(&degI[dst[i]], 1.0f);
    }
}

__global__ void k_rsqrt_clip(float* a, float* b, int n) {
    int i = blockIdx.x * blockDim.x + threadIdx.x;
    if (i < n) {
        a[i] = rsqrtf(fmaxf(a[i], 1.0f));
        b[i] = rsqrtf(fmaxf(b[i], 1.0f));
    }
}

// fp32 -> bf16 with optional per-row scale (row = i / ld)
__global__ void k_to_bf16(const float* src, const float* rowscale,
                          unsigned short* dst, size_t n, unsigned ld) {
    size_t i = (size_t)blockIdx.x * blockDim.x + threadIdx.x;
    if (i < n) {
        float v = src[i];
        if (rowscale) v *= rowscale[i / ld];
        dst[i] = f2bf(v);
    }
}

// pack fp32 weight [K x NcValid] (row-major, leading dim ldw) into CDNA5
// WMMA-B lane layout: per (32K x 16N) tile, 32 lanes x 8 dwords (K pairs).
__global__ void k_pack_b(const float* W, int ldw, int K, int NcDst, int NcValid,
                         unsigned* dst) {
    int ntN = NcDst >> 4;
    int tid = blockIdx.x * blockDim.x + threadIdx.x;
    int total = (K >> 5) * ntN * 256;
    if (tid >= total) return;
    int v    = tid & 7;
    int lane = (tid >> 3) & 31;
    int tile = tid >> 8;
    int ct   = tile % ntN;
    int kt   = tile / ntN;
    int half = lane >> 4;
    int n    = ct * 16 + (lane & 15);
    int k    = kt * 32 + (v >> 2) * 16 + (v & 3) * 2 + half * 8;
    float a0 = (n < NcValid) ? W[(size_t)k * ldw + n]       : 0.0f;
    float a1 = (n < NcValid) ? W[(size_t)(k + 1) * ldw + n] : 0.0f;
    dst[tid] = (unsigned)f2bf(a0) | ((unsigned)f2bf(a1) << 16);
}

// ---------------- edge message scatter (HBM/atomic bound) ----------------
// one wave32 per edge: 32 lanes x float4 = 128 features
__global__ __launch_bounds__(256)
void k_scatter(const float* __restrict__ x, const float* __restrict__ isout,
               const int* __restrict__ src, const int* __restrict__ dst,
               float* __restrict__ agg, int e) {
    int edge = blockIdx.x * 8 + threadIdx.y;
    if (edge >= e) return;
    int s = src[edge], d = dst[edge];
    float sc = isout[s];
    v4f val = ((const v4f*)(x + (size_t)s * HH))[threadIdx.x];
    float* ar = agg + (size_t)d * HH + threadIdx.x * 4;
    unsafeAtomicAdd(ar + 0, val.x * sc);
    unsafeAtomicAdd(ar + 1, val.y * sc);
    unsafeAtomicAdd(ar + 2, val.z * sc);
    unsafeAtomicAdd(ar + 3, val.w * sc);
}

// ---------------- WMMA bf16 GEMM ----------------
// One wave owns one 16-col tile: B K-strip pinned in registers (NTK tiles),
// bias value pinned in a register, then streams RT row tiles.
// All shape constants are template params so epilogue addresses fold into
// immediate store offsets. K = NTK*32 = lda.
// NC: full output width; LDC/NCSTORE: fp32 store ld + col mask; HASB: bf16
// store (ld = NC); ROWADD: += rowadd[row/NPG][col] (NPG literal divisor).
template<int NTK, int NC, int RT, bool ROWADD, bool RELU,
         bool HASF, int LDC, int NCSTORE, bool HASB>
__global__ __launch_bounds__(256)
void k_gemm(const unsigned short* __restrict__ A,
            const unsigned* __restrict__ Bpk,
            const float* __restrict__ bias,
            const float* __restrict__ rowadd,
            float* __restrict__ Cf,
            unsigned short* __restrict__ Cb) {
    constexpr int ntN = NC >> 4;
    constexpr int lda = NTK * 32;
    int ct = blockIdx.y * 8 + threadIdx.y;
    if (ntN < 8 || blockIdx.y) { if (ct >= ntN) return; }
    int lane = threadIdx.x;
    int m    = lane & 15;
    int half = lane >> 4;
    int col  = ct * 16 + m;

    // pin B K-strip + bias in registers (shared across all RT row tiles)
    ABTile b[NTK];
    #pragma unroll
    for (int kt = 0; kt < NTK; ++kt) {
        const v4u* bp = (const v4u*)(Bpk + ((size_t)(kt * ntN + ct) * 32 + lane) * 8);
        b[kt].q[0] = bp[0];
        b[kt].q[1] = bp[1];
    }
    float bval = bias[col];

    int row0 = blockIdx.x * RT * 16;
    const unsigned short* arow = A + (size_t)(row0 + m) * lda + half * 8;

    #pragma unroll 1
    for (int i = 0; i < RT; ++i) {
        const unsigned short* ap = arow + (size_t)i * 16 * lda;
        if (i + 1 < RT)                             // global_prefetch_b8
            __builtin_prefetch(ap + (size_t)16 * lda, 0, 1);
        ABTile a[NTK];
        #pragma unroll
        for (int kt = 0; kt < NTK; ++kt) {
            a[kt].q[0] = *(const v4u*)(ap + kt * 32);
            a[kt].q[1] = *(const v4u*)(ap + kt * 32 + 16);
        }
        v8f acc = {};
        #pragma unroll
        for (int kt = 0; kt < NTK; ++kt)
            acc = __builtin_amdgcn_wmma_f32_16x16x32_bf16(
                      false, a[kt].v, false, b[kt].v, (short)0, acc, false, false);

        int rowb = row0 + i * 16 + half * 8;        // rows rowb .. rowb+7
        float*          cf = HASF ? Cf + (size_t)rowb * LDC + col : nullptr;
        unsigned short* cb = HASB ? Cb + (size_t)rowb * NC  + col : nullptr;
        #pragma unroll
        for (int r = 0; r < 8; ++r) {
            float val = acc[r] + bval;
            if (ROWADD) val += rowadd[(size_t)((unsigned)(rowb + r) / NPG) * NC + col];
            if (RELU) val = fmaxf(val, 0.0f);
            if (HASF) { if (NCSTORE == NC || col < NCSTORE) cf[(size_t)r * LDC] = val; }
            if (HASB) cb[(size_t)r * NC] = f2bf(val);
        }
    }
}

// ---------------- batchnorm over nodes ----------------
__global__ __launch_bounds__(256)
void k_bn_stats(const float* __restrict__ h, float* __restrict__ sums,
                int rows, int rpb) {
    int c   = threadIdx.x & 127;
    int grp = threadIdx.x >> 7;
    int r0 = blockIdx.x * rpb;
    int r1 = min(rows, r0 + rpb);
    float s = 0.0f, q = 0.0f;
    for (int r = r0 + grp; r < r1; r += 2) {
        float v = h[(size_t)r * HH + c];
        s += v; q += v * v;
    }
    __shared__ float ls[256];
    ls[threadIdx.x] = s; __syncthreads();
    if (threadIdx.x < 128) unsafeAtomicAdd(&sums[c], ls[threadIdx.x] + ls[threadIdx.x + 128]);
    __syncthreads();
    ls[threadIdx.x] = q; __syncthreads();
    if (threadIdx.x < 128) unsafeAtomicAdd(&sums[128 + c], ls[threadIdx.x] + ls[threadIdx.x + 128]);
}

__global__ void k_bn_finalize(const float* sums, float* musig) {
    int c = threadIdx.x;
    if (c < 128) {
        float mu  = sums[c] / (float)NN;
        float var = sums[128 + c] / (float)NN - mu * mu;
        musig[c]       = mu;
        musig[128 + c] = rsqrtf(var + EPSI);
    }
}

// x = x + relu(gamma*(h-mu)*invstd + beta)
__global__ void k_bn_apply(const float* __restrict__ h, const float* __restrict__ musig,
                           const float* __restrict__ gamma, const float* __restrict__ beta,
                           float* __restrict__ x, size_t n) {
    size_t i = (size_t)blockIdx.x * blockDim.x + threadIdx.x;
    if (i < n) {
        int c = (int)(i & 127);
        float v = gamma[c] * (h[i] - musig[c]) * musig[128 + c] + beta[c];
        x[i] += fmaxf(v, 0.0f);
    }
}

// ---------------- pooling + graph head ----------------
__global__ __launch_bounds__(128)
void k_pool(const float* __restrict__ x, float* __restrict__ hg) {
    int g = blockIdx.x, c = threadIdx.x;
    float s = 0.0f, mx = -3.4e38f;
    const float* base = x + (size_t)g * NPG * HH + c;
    for (int i = 0; i < NPG; ++i) {
        float v = base[(size_t)i * HH];
        s += v; mx = fmaxf(mx, v);
    }
    hg[(size_t)g * 256 + c]       = s / (float)NPG;
    hg[(size_t)g * 256 + 128 + c] = mx;
}

__global__ __launch_bounds__(256)
void k_graph_head(const float* __restrict__ hg,
                  const float* gW1, const float* gb1,
                  const float* gW2, const float* gb2,
                  const float* gW3, const float* gb3,
                  const float* nW1, float* __restrict__ g_out,
                  float* __restrict__ g1) {
    int g = blockIdx.x, tid = threadIdx.x;
    __shared__ float buf[256];
    __shared__ float t1[128];
    __shared__ float t2[64];
    buf[tid] = hg[(size_t)g * 256 + tid];
    __syncthreads();
    if (tid < 192) {                       // g1 = hg @ nW1[:256,:]   (node-head rowadd)
        float s = 0.0f;
        for (int k = 0; k < 256; ++k) s += buf[k] * nW1[(size_t)k * 192 + tid];
        g1[(size_t)g * 192 + tid] = s;
    }
    if (tid < 128) {
        float s = gb1[tid];
        for (int k = 0; k < 256; ++k) s += buf[k] * gW1[(size_t)k * 128 + tid];
        t1[tid] = fmaxf(s, 0.0f);
    }
    __syncthreads();
    if (tid < 64) {
        float s = gb2[tid];
        for (int k = 0; k < 128; ++k) s += t1[k] * gW2[(size_t)k * 64 + tid];
        t2[tid] = fmaxf(s, 0.0f);
    }
    __syncthreads();
    if (tid < CC) {
        float s = gb3[tid];
        for (int k = 0; k < 64; ++k) s += t2[k] * gW3[(size_t)k * CC + tid];
        g_out[(size_t)g * CC + tid] = s;
    }
}

__global__ void k_labels(const int* labels, float* out, int n) {
    int i = blockIdx.x * blockDim.x + threadIdx.x;
    if (i < n) out[i] = (float)labels[i / NPG];
}

// ---------------- launch ----------------
extern "C" void kernel_launch(void* const* d_in, const int* in_sizes, int n_in,
                              void* d_out, int out_size, void* d_ws, size_t ws_size,
                              hipStream_t stream) {
    const float* feature = (const float*)d_in[0];
    const int*   esrc    = (const int*)d_in[1];
    const int*   edst    = (const int*)d_in[2];
    const int*   labels  = (const int*)d_in[3];
    const float* W_emb   = (const float*)d_in[4];
    const float* b_emb   = (const float*)d_in[5];
    const float* gcnW    = (const float*)d_in[6];
    const float* gcnB    = (const float*)d_in[7];
    const float* gcnG    = (const float*)d_in[8];
    const float* gcnBe   = (const float*)d_in[9];
    const float* gW1 = (const float*)d_in[10]; const float* gb1 = (const float*)d_in[11];
    const float* gW2 = (const float*)d_in[12]; const float* gb2 = (const float*)d_in[13];
    const float* gW3 = (const float*)d_in[14]; const float* gb3 = (const float*)d_in[15];
    const float* nW1 = (const float*)d_in[16]; const float* nb1 = (const float*)d_in[17];
    const float* nW2 = (const float*)d_in[18]; const float* nb2 = (const float*)d_in[19];
    const float* nW3 = (const float*)d_in[20]; const float* nb3 = (const float*)d_in[21];

    // output regions
    float* x_out   = (float*)d_out;                       // [N,128]
    float* xg_out  = x_out + (size_t)NN * HH;             // [N,10]
    float* g_out   = xg_out + (size_t)NN * CC;            // [B,10]
    float* lab_out = g_out + (size_t)BB * CC;             // [N]

    // workspace carving (256B aligned)
    char* wsp = (char*)d_ws; size_t off = 0;
    auto carve = [&](size_t bytes) {
        void* p = wsp + off;
        off = (off + bytes + 255) & ~(size_t)255;
        return p;
    };
    float* isout  = (float*)carve((size_t)NN * 4);
    float* isin   = (float*)carve((size_t)NN * 4);
    float* hbuf   = (float*)carve((size_t)NN * HH * 4);       // agg, then GEMM out
    unsigned short* abuf = (unsigned short*)carve((size_t)NN * HH  * 2);
    unsigned short* h1b  = (unsigned short*)carve((size_t)NN * 192 * 2);
    unsigned short* h2b  = (unsigned short*)carve((size_t)NN * 96  * 2);
    float* hg     = (float*)carve((size_t)BB * 256 * 4);
    float* g1     = (float*)carve((size_t)BB * 192 * 4);
    float* sums   = (float*)carve(256 * 4);
    float* musig  = (float*)carve(256 * 4);
    float* nb3pad = (float*)carve(16 * 4);
    unsigned* pk_emb = (unsigned*)carve(8192 * 4);
    unsigned* pk_gcn = (unsigned*)carve(3 * 8192 * 4);
    unsigned* pk_n1  = (unsigned*)carve(12288 * 4);
    unsigned* pk_n2  = (unsigned*)carve(9216 * 4);
    unsigned* pk_n3  = (unsigned*)carve(768 * 4);
    (void)ws_size; (void)n_in; (void)in_sizes; (void)out_size;

    const int RT  = 16;
    const int MTB = NN / (16 * RT);                       // 391 row blocks (exact)
    auto g1d = [](size_t n) { return dim3((unsigned)((n + 255) / 256)); };

    // degrees -> rsqrt(clip)
    k_fill<<<g1d(NN), 256, 0, stream>>>(isout, NN, 0.0f);
    k_fill<<<g1d(NN), 256, 0, stream>>>(isin,  NN, 0.0f);
    k_degrees<<<g1d(EE), 256, 0, stream>>>(esrc, edst, isout, isin, EE);
    k_rsqrt_clip<<<g1d(NN), 256, 0, stream>>>(isout, isin, NN);

    // pack all WMMA B operands to bf16 lane layout; pad nb3 to 16 cols
    k_pack_b<<<g1d(8192), 256, 0, stream>>>(W_emb, HH, HH, HH, HH, pk_emb);
    for (int i = 0; i < LL; ++i)
        k_pack_b<<<g1d(8192), 256, 0, stream>>>(gcnW + (size_t)i * HH * HH, HH, HH, HH, HH,
                                                pk_gcn + (size_t)i * 8192);
    k_pack_b<<<g1d(12288), 256, 0, stream>>>(nW1 + (size_t)256 * 192, 192, 128, 192, 192, pk_n1);
    k_pack_b<<<g1d(9216),  256, 0, stream>>>(nW2, 96, 192, 96, 96, pk_n2);
    k_pack_b<<<g1d(768),   256, 0, stream>>>(nW3, CC, 96, 16, CC, pk_n3);
    k_pad16<<<dim3(1), 16, 0, stream>>>(nb3, nb3pad, CC);

    // embedding: x = feature @ W_emb + b_emb
    k_to_bf16<<<g1d((size_t)NN * HH), 256, 0, stream>>>(feature, nullptr, abuf,
                                                        (size_t)NN * HH, HH);
    k_gemm<4, 128, RT, false, false, true, 128, 128, false>
        <<<dim3(MTB, 1), dim3(32, 8), 0, stream>>>(
        abuf, pk_emb, b_emb, nullptr, x_out, nullptr);

    // GCN layers
    for (int i = 0; i < LL; ++i) {
        k_fill<<<g1d((size_t)NN * HH), 256, 0, stream>>>(hbuf, (size_t)NN * HH, 0.0f);
        k_scatter<<<dim3((EE + 7) / 8), dim3(32, 8), 0, stream>>>(x_out, isout, esrc, edst, hbuf, EE);
        k_to_bf16<<<g1d((size_t)NN * HH), 256, 0, stream>>>(hbuf, isin, abuf, (size_t)NN * HH, HH);
        k_gemm<4, 128, RT, false, false, true, 128, 128, false>
            <<<dim3(MTB, 1), dim3(32, 8), 0, stream>>>(
            abuf, pk_gcn + (size_t)i * 8192, gcnB + (size_t)i * HH, nullptr, hbuf, nullptr);
        k_fill<<<dim3(1), 256, 0, stream>>>(sums, 256, 0.0f);
        k_bn_stats<<<dim3(391), 256, 0, stream>>>(hbuf, sums, NN, 256);
        k_bn_finalize<<<dim3(1), 128, 0, stream>>>(sums, musig);
        k_bn_apply<<<g1d((size_t)NN * HH), 256, 0, stream>>>(hbuf, musig,
            gcnG + (size_t)i * HH, gcnBe + (size_t)i * HH, x_out, (size_t)NN * HH);
    }

    // pooling + graph head (also produces g1 = hg @ nW1[:256] for node head)
    k_pool<<<dim3(BB), 128, 0, stream>>>(x_out, hg);
    k_graph_head<<<dim3(BB), 256, 0, stream>>>(hg, gW1, gb1, gW2, gb2, gW3, gb3,
                                               nW1, g_out, g1);

    // node head: h1 = relu(x@nW1[256:] + g1[seg] + nb1); h2 = relu(h1@nW2+nb2); xg = h2@nW3+nb3
    k_to_bf16<<<g1d((size_t)NN * HH), 256, 0, stream>>>(x_out, nullptr, abuf, (size_t)NN * HH, HH);
    k_gemm<4, 192, RT, true, true, false, 1, 0, true>
        <<<dim3(MTB, 2), dim3(32, 8), 0, stream>>>(
        abuf, pk_n1, nb1, g1, nullptr, h1b);
    k_gemm<6, 96, RT, false, true, false, 1, 0, true>
        <<<dim3(MTB, 1), dim3(32, 8), 0, stream>>>(
        h1b, pk_n2, nb2, nullptr, nullptr, h2b);
    k_gemm<3, 16, RT, false, false, true, 10, 10, false>
        <<<dim3(MTB, 1), dim3(32, 8), 0, stream>>>(
        h2b, pk_n3, nb3pad, nullptr, xg_out, nullptr);

    // node labels
    k_labels<<<g1d(NN), 256, 0, stream>>>(labels, lab_out, NN);
}
